// STEM_51238959841965
// MI455X (gfx1250) — matrix operand order repaired and checked by reading, not compiled, not run on hardware
//
#include <hip/hip_runtime.h>
#include <hip/hip_bf16.h>
#include <math.h>

// ---------------------------------------------------------------------------
// MMoE forward for MI455X (gfx1250, wave32).
// Tower GEMM: [2048,128] x [128,N] bf16 WMMA, fp32 accumulate.
// B tile staged pre-transposed in LDS once per block; compiler hoists all 16
// B fragments into VGPRs, inner M-loop = 8x global_load_b128 (A) + 16 WMMA +
// stores. Bias and store masks hoisted out of the M-loop so stores issue
// without a dependent load. GEMM is output-store bound (~827 MB / 23.3 TB/s).
// ---------------------------------------------------------------------------

typedef __attribute__((ext_vector_type(16))) __bf16  v16bf;
typedef __attribute__((ext_vector_type(8)))  float   v8f;

#define B_     2048
#define L_     50
#define EMB_   128
#define IND_   256   // 2*EMB
#define HID_   128
#define NEXP_  6
#define NTASK_ 2
#define NITEM_ 100000
#define NCAT_  1000

__device__ __forceinline__ unsigned short f2bf(float f) {
  union { float f; unsigned u; } v; v.f = f;
  unsigned r = v.u + 0x7FFFu + ((v.u >> 16) & 1u);  // round-to-nearest-even
  return (unsigned short)(r >> 16);
}

// --------------------------- 1) embedding mean-pool -------------------------
__global__ void pool_kernel(const int* __restrict__ items,
                            const int* __restrict__ cats,
                            const float* __restrict__ item_emb,
                            const float* __restrict__ cat_emb,
                            float* __restrict__ xout) {
  int b = blockIdx.x;
  int t = threadIdx.x;
  bool isItem = t < EMB_;
  int d = isItem ? t : t - EMB_;
  const int*   idxp = isItem ? (items + (size_t)b * L_) : (cats + (size_t)b * L_);
  const float* emb  = isItem ? item_emb : cat_emb;
  float acc = 0.f;
  #pragma unroll 5
  for (int l = 0; l < L_; ++l)
    acc += emb[(size_t)idxp[l] * EMB_ + d];
  xout[(size_t)b * IND_ + t] = acc * (1.0f / (float)L_);
}

// --------------------------- 2) fp32 -> bf16 convert ------------------------
__global__ void f32_to_bf16(const float* __restrict__ src,
                            unsigned short* __restrict__ dst, int n) {
  int i = blockIdx.x * blockDim.x + threadIdx.x;
  int stride = gridDim.x * blockDim.x;
  for (; i < n; i += stride) dst[i] = f2bf(src[i]);
}

// --------------------- 3) experts + gates + softmax mixture -----------------
__global__ void moe_kernel(const float* __restrict__ x,
                           const float* __restrict__ eW,   // [6,256,128]
                           const float* __restrict__ eB,   // [6,128]
                           const float* __restrict__ gW,   // [2,256,6]
                           const float* __restrict__ gB,   // [2,6]
                           unsigned short* __restrict__ mixbf) {
  __shared__ float xs[IND_];
  __shared__ float hs[NEXP_ * HID_];
  __shared__ float gl[NTASK_ * NEXP_];
  int b = blockIdx.x, t = threadIdx.x;
  xs[t] = x[(size_t)b * IND_ + t];
  __syncthreads();

  for (int o = t; o < NEXP_ * HID_; o += blockDim.x) {
    int e = o >> 7, hd = o & (HID_ - 1);
    const float* W = eW + (size_t)e * IND_ * HID_ + hd;
    float a = 0.f;
    #pragma unroll 8
    for (int k = 0; k < IND_; ++k) a += xs[k] * W[(size_t)k * HID_];
    a += eB[e * HID_ + hd];
    hs[o] = fmaxf(a, 0.f);
  }
  if (t < NTASK_ * NEXP_) {
    int task = t / NEXP_, g = t % NEXP_;
    const float* W = gW + (size_t)task * IND_ * NEXP_ + g;
    float a = 0.f;
    for (int k = 0; k < IND_; ++k) a += xs[k] * W[(size_t)k * NEXP_];
    gl[t] = a + gB[t];
  }
  __syncthreads();

  {
    int task = t >> 7, hd = t & (HID_ - 1);
    float m = -1e30f;
    for (int g = 0; g < NEXP_; ++g) m = fmaxf(m, gl[task * NEXP_ + g]);
    float wgt[NEXP_], s = 0.f;
    for (int g = 0; g < NEXP_; ++g) { wgt[g] = __expf(gl[task * NEXP_ + g] - m); s += wgt[g]; }
    float inv = 1.f / s, v = 0.f;
    for (int g = 0; g < NEXP_; ++g) v += wgt[g] * inv * hs[g * HID_ + hd];
    mixbf[((size_t)task * B_ + b) * HID_ + hd] = f2bf(v);
  }
}

// ------------------------ 4) tower GEMM via bf16 WMMA -----------------------
// Block: 256 threads = 8 waves arranged 4(M) x 2(N-groups of 64).
// Block tile: 64 M x 128 N per iteration; M-loop covers 1024 rows (grid.y=2).
// B tile [128 cols][128 k] staged transposed in LDS (row pad -> 136 ushorts,
// 272 B = 17*16 keeps b128 alignment). Fragment layouts per ISA 7.12.2:
//   A: lane holds row m; elem i -> K = kb + 8*half + (i&7) + 16*(i>>3)
//   B: lane holds col (lane&15); elem i -> K = kb + 8*half + (i&7) + 16*(i>>3)
//   D: VGPR r -> M = mt + r + 8*half, col = lane&15
__global__ __launch_bounds__(256) void tower_wmma_kernel(
    const unsigned short* __restrict__ A,   // [2048,128] bf16 row-major
    const unsigned short* __restrict__ W,   // [128,N]    bf16 row-major
    const float* __restrict__ bias,         // [N]
    float* __restrict__ out,                // [2048,N] fp32
    int N) {
  const int K = 128;
  __shared__ unsigned short tile[128][136];   // [local col][k], padded row

  const int t  = threadIdx.x;
  const int n0 = blockIdx.x * 128;

  // ---- stage B tile transposed: coalesced b32 row reads, b16 LDS writes ----
  {
    int nn = (t & 63) * 2;                    // local column pair
    int k0 = t >> 6;                          // 0..3
    int cp = n0 + nn;
    if (cp > N - 2) cp = N - 2;               // N is even; clamp pair (dup ok)
    for (int k = k0; k < K; k += 4) {
      unsigned int v = *(const unsigned int*)(W + (size_t)k * N + cp);
      tile[nn][k]     = (unsigned short)(v & 0xFFFFu);
      tile[nn + 1][k] = (unsigned short)(v >> 16);
    }
  }
  __syncthreads();

  const int lane   = t & 31;
  const int wave   = t >> 5;                  // 0..7
  const int lane16 = lane & 15;
  const int half   = lane >> 4;
  const int mrow   = wave & 3;                // 0..3  -> M sub-tile
  const int nlocal = (wave >> 2) * 64;        // 0 or 64 -> N sub-group
  const int mbase  = blockIdx.y * 1024;

  // ---- hoist per-column epilogue state out of the M-loop ----
  int   colv[4];
  bool  okv[4];
  float bv[4];
  #pragma unroll
  for (int j = 0; j < 4; ++j) {
    int col = n0 + nlocal + j * 16 + lane16;
    colv[j] = col;
    okv[j]  = (col < N);
    bv[j]   = bias[okv[j] ? col : (N - 1)];   // clamped: no divergence, no OOB
  }

  for (int m0 = mbase; m0 < mbase + 1024; m0 += 64) {
    const int mt = m0 + mrow * 16;

    v8f acc[4];
    #pragma unroll
    for (int j = 0; j < 4; ++j)
      #pragma unroll
      for (int r = 0; r < 8; ++r) acc[j][r] = 0.f;

    const unsigned short* Arow = A + (size_t)(mt + lane16) * K + half * 8;

    #pragma unroll
    for (int kb = 0; kb < K; kb += 32) {
      union { v16bf v; uint4 q[2]; } af;
      af.q[0] = *(const uint4*)(Arow + kb);        // K = kb+8*half +{0..7}
      af.q[1] = *(const uint4*)(Arow + kb + 16);   // K = kb+8*half+16+{0..7}

      #pragma unroll
      for (int j = 0; j < 4; ++j) {
        int c = nlocal + j * 16 + lane16;          // local column
        union { v16bf v; uint4 q[2]; } bfv;
        const uint4* p = (const uint4*)&tile[c][kb + half * 8];
        bfv.q[0] = p[0];                           // K = kb+8*half +{0..7}
        bfv.q[1] = p[2];                           // K = kb+8*half+16+{0..7}
        acc[j] = __builtin_amdgcn_wmma_f32_16x16x32_bf16(
            false, af.v, false, bfv.v, (short)0, acc[j], false, false);
      }
    }

    const float* __restrict__ biasless = bias;  (void)biasless;
    float* orow = out + (size_t)(mt + half * 8) * N;
    #pragma unroll
    for (int j = 0; j < 4; ++j) {
      if (okv[j]) {
        #pragma unroll
        for (int r = 0; r < 8; ++r)
          orow[(size_t)r * N + colv[j]] = acc[j][r] + bv[j];
      }
    }
  }
}

// ---------------------------------------------------------------------------
extern "C" void kernel_launch(void* const* d_in, const int* in_sizes, int n_in,
                              void* d_out, int out_size, void* d_ws, size_t ws_size,
                              hipStream_t stream) {
  (void)in_sizes; (void)n_in; (void)out_size; (void)ws_size;

  const int*   items    = (const int*)d_in[0];
  const int*   cats     = (const int*)d_in[1];
  const float* item_emb = (const float*)d_in[2];
  const float* cat_emb  = (const float*)d_in[3];
  const float* eW       = (const float*)d_in[4];
  const float* eB       = (const float*)d_in[5];
  const float* gW       = (const float*)d_in[6];
  const float* gB       = (const float*)d_in[7];
  const float* tiW      = (const float*)d_in[8];
  const float* tiB      = (const float*)d_in[9];
  const float* tcW      = (const float*)d_in[10];
  const float* tcB      = (const float*)d_in[11];

  // workspace carve-out (256B-aligned offsets), ~29 MB total
  char* w = (char*)d_ws;
  float*          xpool = (float*)w;                                  // 2,097,152 B
  unsigned short* mixbf = (unsigned short*)(w + (size_t)2097152);     // 1,048,576 B
  unsigned short* Wibf  = (unsigned short*)(w + (size_t)3145728);     // 25,600,000 B
  unsigned short* Wcbf  = (unsigned short*)(w + (size_t)28745728);    // 256,000 B

  pool_kernel<<<B_, 256, 0, stream>>>(items, cats, item_emb, cat_emb, xpool);
  f32_to_bf16<<<2048, 256, 0, stream>>>(tiW, Wibf, 128 * NITEM_);
  f32_to_bf16<<<64,   256, 0, stream>>>(tcW, Wcbf, 128 * NCAT_);
  moe_kernel<<<B_, 256, 0, stream>>>(xpool, eW, eB, gW, gB, mixbf);

  float* item_logits = (float*)d_out;
  float* cat_logits  = (float*)d_out + (size_t)B_ * NITEM_;

  dim3 gi((NITEM_ + 127) / 128, 2);
  tower_wmma_kernel<<<gi, 256, 0, stream>>>(mixbf, Wibf, tiB, item_logits, NITEM_);

  dim3 gc((NCAT_ + 127) / 128, 2);
  tower_wmma_kernel<<<gc, 256, 0, stream>>>(mixbf + (size_t)B_ * HID_, Wcbf, tcB,
                                            cat_logits, NCAT_);
}
